// Model_29119878266971
// MI455X (gfx1250) — matrix operand order repaired and checked by reading, not compiled, not run on hardware
//
#include <hip/hip_runtime.h>
#include <hip/hip_bf16.h>
#include <math.h>
#include <stdint.h>

// ---------------- model dimensions ----------------
#define Bsz  8
#define Nn   25
#define Ls   192
#define PREDn 48
#define Cc   4
#define Jj   3
#define Dd   128
#define DFFn 256
#define Hh   4
#define Ee   800
#define BNn  (Bsz*Nn)        // 200
#define SEQn (Ls-PREDn)      // 144
#define LFn  97              // rfft bins (L/2+1)
#define LXn  (2*LFn)         // 194

#define TOKE ((long)Ee*Ls*Dd)       // 19,660,800
#define TOKN ((long)BNn*Ls*Dd)      //  4,915,200
#define TOKF ((long)BNn*LXn*Dd)     //  4,966,400
#define FFNE ((long)Ee*Ls*DFFn)     // 39,321,600

#define PIF 3.14159265358979323846f

typedef __bf16 bf16;
typedef __attribute__((ext_vector_type(16))) __bf16 v16bf;
typedef __attribute__((ext_vector_type(8)))  __bf16 bf8;
typedef __attribute__((ext_vector_type(8)))  float  v8f;

__device__ __forceinline__ float gelu_f(float x){
    float x3 = x*x*x;
    return 0.5f*x*(1.0f + tanhf(0.7978845608028654f*(x + 0.044715f*x3)));
}

__device__ __forceinline__ bf8 cvt8(float4 a, float4 b){
    bf8 v;
    v[0]=(bf16)a.x; v[1]=(bf16)a.y; v[2]=(bf16)a.z; v[3]=(bf16)a.w;
    v[4]=(bf16)b.x; v[5]=(bf16)b.y; v[6]=(bf16)b.z; v[7]=(bf16)b.w;
    return v;
}

// ---------------- utility kernels ----------------
__global__ void zero_kernel(float* __restrict__ p, long n){
    long i = (long)blockIdx.x*256 + threadIdx.x;
    if (i < n) p[i] = 0.0f;
}

__global__ void cvt_bf16_kernel(bf16* __restrict__ dst, const float* __restrict__ src, int n){
    int i = blockIdx.x*256 + threadIdx.x;
    if (i < n) dst[i] = (bf16)src[i];
}

// ---------------- wavelet decomposition ----------------
__global__ __launch_bounds__(256)
void wavelet_kernel(const float* __restrict__ nodes, float* __restrict__ freq, float* __restrict__ trend){
    __shared__ float a[SEQn][Cc];
    __shared__ float mn[Cc];
    int bn = blockIdx.x;
    int tid = threadIdx.x;
    for (int e = tid; e < SEQn*Cc; e += 256){
        int l = e / Cc, ch = e % Cc;
        a[l][ch] = nodes[((size_t)bn*Ls + l)*Cc + ch];
    }
    __syncthreads();
    for (int j = 0; j < Jj; j++){
        int shift = 1 << j;
        float na_val[3]; int it = 0;
        for (int e = tid; e < SEQn*Cc; e += 256){
            int l = e / Cc, ch = e % Cc;
            float av = a[l][ch];
            float sh = a[(l - shift + SEQn) % SEQn][ch];
            freq[((size_t)bn*Ls + l)*(Jj*Cc) + j*Cc + ch] = (av - sh)*0.5f;
            na_val[it++] = (av + sh)*0.5f;
        }
        __syncthreads();
        it = 0;
        for (int e = tid; e < SEQn*Cc; e += 256){
            int l = e / Cc, ch = e % Cc;
            a[l][ch] = na_val[it++];
        }
        __syncthreads();
    }
    if (tid < Cc){
        float s = 0.0f;
        for (int l = 0; l < SEQn; l++) s += a[l][tid];
        mn[tid] = s / (float)SEQn;
    }
    __syncthreads();
    for (int e = tid; e < Ls*Cc; e += 256){
        int l = e / Cc, ch = e % Cc;
        trend[((size_t)bn*Ls + l)*Cc + ch] = (l < SEQn) ? a[l][ch] : mn[ch];
    }
    for (int e = tid; e < (Ls-SEQn)*(Jj*Cc); e += 256){
        int l = SEQn + e/(Jj*Cc), ch = e % (Jj*Cc);
        freq[((size_t)bn*Ls + l)*(Jj*Cc) + ch] = 0.0f;
    }
}

// ---------------- embeddings (tiny K, direct VALU conv) ----------------
__global__ __launch_bounds__(128)
void embed_freq_kernel(float* __restrict__ xf, const float* __restrict__ freq, const float* __restrict__ w){
    int bl = blockIdx.x; int l = bl % Ls; int bn = bl / Ls;
    int d = threadIdx.x;
    float acc = 0.0f;
    #pragma unroll
    for (int i = 0; i < 3; i++){
        int ls = (l + i - 1 + Ls) % Ls;   // circular pad 1
        const float* fp = freq + ((size_t)bn*Ls + ls)*(Jj*Cc);
        #pragma unroll
        for (int cin = 0; cin < Jj*Cc; cin++)
            acc += fp[cin] * w[((size_t)i*(Jj*Cc)+cin)*Dd + d];
    }
    xf[((size_t)bn*Ls + l)*Dd + d] = acc;
}

__global__ __launch_bounds__(128)
void embed_trend_kernel(float* __restrict__ xt, const float* __restrict__ trend,
                        const float* __restrict__ w, const float* __restrict__ xmark,
                        const float* __restrict__ wtemp){
    int bl = blockIdx.x; int l = bl % Ls; int bn = bl / Ls;
    int b = bn / Nn;
    int d = threadIdx.x;
    float acc = 0.0f;
    #pragma unroll
    for (int i = 0; i < 3; i++){
        int ls = (l + i - 1 + Ls) % Ls;
        const float* tp = trend + ((size_t)bn*Ls + ls)*Cc;
        #pragma unroll
        for (int cin = 0; cin < Cc; cin++)
            acc += tp[cin] * w[((size_t)i*Cc+cin)*Dd + d];
    }
    const float* xm = xmark + ((size_t)b*Ls + l)*4;
    #pragma unroll
    for (int k = 0; k < 4; k++) acc += xm[k]*wtemp[(size_t)k*Dd + d];
    float pos = (float)l;
    if (d & 1){
        float ang = pos * powf(10000.0f, -(float)(d-1)/(float)Dd);
        acc += cosf(ang);
    } else {
        float ang = pos * powf(10000.0f, -(float)d/(float)Dd);
        acc += sinf(ang);
    }
    xt[((size_t)bn*Ls + l)*Dd + d] = acc;
}

// ---------------- edge init / graph ops ----------------
__global__ void edge_init_kernel(float* __restrict__ eb, const float* __restrict__ edges,
                                 const float* __restrict__ we, const float* __restrict__ be){
    long i = (long)blockIdx.x*256 + threadIdx.x;
    if (i >= TOKE) return;
    int d = (int)(i % Dd); long r = i / Dd; int e = (int)(r / Ls);
    eb[i] = edges[e]*we[d] + be[d];
}

__global__ void count_kernel(float* __restrict__ cnt, const int* __restrict__ rcv){
    int e = blockIdx.x*256 + threadIdx.x;
    if (e < Ee) atomicAdd(&cnt[rcv[e]], 1.0f);
}

__global__ void gather_ein_kernel(float* __restrict__ ein, const float* __restrict__ eb,
                                  const float* __restrict__ xt, const int* __restrict__ snd,
                                  const int* __restrict__ rcv){
    long i = (long)blockIdx.x*256 + threadIdx.x;
    if (i >= TOKE) return;
    int d = (int)(i % Dd); long r = i / Dd; int l = (int)(r % Ls); int e = (int)(r / Ls);
    int s = snd[e], t = rcv[e];
    ein[i] = eb[i] + xt[((size_t)s*Ls + l)*Dd + d] + xt[((size_t)t*Ls + l)*Dd + d];
}

__global__ void agg_kernel(float* __restrict__ tin, const float* __restrict__ eout,
                           const int* __restrict__ rcv){
    long i = (long)blockIdx.x*256 + threadIdx.x;
    if (i >= TOKE) return;
    int d = (int)(i % Dd); long r = i / Dd; int l = (int)(r % Ls); int e = (int)(r / Ls);
    atomicAdd(&tin[((size_t)rcv[e]*Ls + l)*Dd + d], eout[i]);
}

__global__ void tin_finish_kernel(float* __restrict__ tin, const float* __restrict__ xt,
                                  const float* __restrict__ cnt){
    long i = (long)blockIdx.x*256 + threadIdx.x;
    if (i >= TOKN) return;
    int bn = (int)(i / ((long)Ls*Dd));
    tin[i] = xt[i] + tin[i] / fmaxf(cnt[bn], 1.0f);
}

// ---------------- WMMA GEMM: Y[M,N] = act(A[M,K]*Wbf16[K,N] (+R)); optional bf16 out ----------------
__global__ __launch_bounds__(256)
void gemm_bf16_kernel(float* __restrict__ Y, bf16* __restrict__ Ybf,
                      const float* __restrict__ A,
                      const bf16* __restrict__ W, const float* __restrict__ R,
                      int M, int K, int N, int act){
    __shared__ __align__(16) bf16 Al[64][32];
    __shared__ __align__(16) bf16 Wl[32][136];   // padded rows: breaks bank clash
    int m0   = blockIdx.x * 64;
    int nB   = blockIdx.y * 128;
    int tid  = threadIdx.x;
    int lane = tid & 31;
    int w    = tid >> 5;
    int wm   = w & 3, wn = w >> 2;
    int rowA  = wm*16 + (lane & 15);
    int khalf = lane >> 4;
    // tile-staging coordinates (vectorized, 16B granularity)
    int ar = tid >> 2;            // 0..63  A-tile row
    int ac = (tid & 3) * 8;       // 0,8,16,24
    int wr = tid >> 3;            // 0..31  W-tile row
    int wc = (tid & 7) * 16;      // 0..112
    int gr = min(m0 + ar, M-1);   // clamped: over-reads are discarded at store
    v8f acc[4];
    #pragma unroll
    for (int a = 0; a < 4; a++)
        #pragma unroll
        for (int r = 0; r < 8; r++) acc[a][r] = 0.0f;

    for (int k0 = 0; k0 < K; k0 += 32){
        // ---- W tile: async global->LDS b128 copies (no VGPR round-trip) ----
        {
            const bf16* wp2 = W + (size_t)(k0 + wr)*N + nB + wc;
            unsigned ldsa = (unsigned)(uintptr_t)&Wl[wr][wc];   // AS(3) byte offset
            asm volatile("global_load_async_to_lds_b128 %0, %1, off"
                         :: "v"(ldsa), "v"(wp2) : "memory");
            asm volatile("global_load_async_to_lds_b128 %0, %1, off offset:16"
                         :: "v"(ldsa), "v"(wp2) : "memory");
        }
        // ---- A tile: two float4 loads -> packed bf16 -> one b128 LDS store ----
        {
            const float* ap = A + (size_t)gr*K + k0 + ac;
            float4 f0 = *(const float4*)ap;
            float4 f1 = *(const float4*)(ap + 4);
            *(bf8*)&Al[ar][ac] = cvt8(f0, f1);
        }
        asm volatile("s_wait_asynccnt 0x0" ::: "memory");
        __syncthreads();
        v16bf af;
        {   // A frag: lane<16 holds K {0..7,16..23}; lane>=16 holds K {8..15,24..31}
            bf8 lo = *(const bf8*)&Al[rowA][8*khalf];
            bf8 hi = *(const bf8*)&Al[rowA][16 + 8*khalf];
            #pragma unroll
            for (int t = 0; t < 8; t++){ af[t] = lo[t]; af[t+8] = hi[t]; }
        }
        #pragma unroll
        for (int a = 0; a < 4; a++){
            int n0 = wn*64 + a*16;
            v16bf bfv;  // B frag: lane%16 = N column, elements = K {8h..8h+7, 16+8h..16+8h+7}
            #pragma unroll
            for (int t = 0; t < 8; t++){ bfv[t] = Wl[8*khalf + t][n0 + (lane & 15)]; bfv[t+8] = Wl[16 + 8*khalf + t][n0 + (lane & 15)]; }
            acc[a] = __builtin_amdgcn_wmma_f32_16x16x32_bf16(false, af, false, bfv,
                                                             (short)0, acc[a], false, false);
        }
        __syncthreads();
    }
    #pragma unroll
    for (int a = 0; a < 4; a++){
        int col = nB + wn*64 + a*16 + (lane & 15);
        #pragma unroll
        for (int r = 0; r < 8; r++){
            int grow = m0 + wm*16 + r + 8*khalf;
            if (grow < M){
                float vv = acc[a][r];
                if (Ybf){
                    Ybf[(size_t)grow*N + col] = (bf16)vv;
                } else {
                    if (R) vv += R[(size_t)grow*N + col];
                    if (act == 1) vv = gelu_f(vv);
                    Y[(size_t)grow*N + col] = vv;
                }
            }
        }
    }
}

// ---------------- fused multi-head attention (one wave per (m,h,qtile)); bf16 inputs ----------------
#define LKPAD 224
__global__ __launch_bounds__(32)
void attn_kernel(float* __restrict__ O, const bf16* __restrict__ Q,
                 const bf16* __restrict__ Kd, const bf16* __restrict__ V,
                 int M, int Lq, int Lk){
    __shared__ __align__(16) float Sl[16][LKPAD];
    __shared__ __align__(16) bf16  Pl[16][LKPAD];
    int QT = (Lq + 15) >> 4;
    int wg = blockIdx.x;
    int qt = wg % QT;
    int mh = wg / QT;
    int h  = mh % Hh;
    int m  = mh / Hh;
    int lane  = threadIdx.x;
    int rloc  = lane & 15;
    int khalf = lane >> 4;
    int q0 = qt * 16;
    const float scale = 0.17677669529663687f;   // 1/sqrt(32)

    // ---- Q fragment: 2 b128 loads, clamped row (over-reads discarded at store) ----
    v16bf aq;
    {
        int row = min(q0 + rloc, Lq-1);
        const bf16* qp = Q + ((size_t)m*Lq + row)*Dd + h*32;
        bf8 lo = *(const bf8*)(qp + 8*khalf);
        bf8 hi = *(const bf8*)(qp + 16 + 8*khalf);
        #pragma unroll
        for (int t = 0; t < 8; t++){ aq[t] = lo[t]; aq[t+8] = hi[t]; }
    }
    int KT = (Lk + 15) >> 4;
    for (int kt = 0; kt < KT; kt++){
        int kk0 = kt*16;
        v16bf bk;   // B = K^T : lane%16 = key column, elements = d {8h..8h+7, 16+8h..16+8h+7}
        {   // clamp key row (softmax never reads cols >= Lk)
            const bf16* kp = Kd + ((size_t)m*Lk + min(kk0 + rloc, Lk-1))*Dd + h*32;
            bf8 lo = *(const bf8*)(kp + 8*khalf);
            bf8 hi = *(const bf8*)(kp + 16 + 8*khalf);
            #pragma unroll
            for (int t = 0; t < 8; t++){ bk[t] = lo[t]; bk[t+8] = hi[t]; }
        }
        v8f c;
        #pragma unroll
        for (int r = 0; r < 8; r++) c[r] = 0.0f;
        c = __builtin_amdgcn_wmma_f32_16x16x32_bf16(false, aq, false, bk, (short)0, c, false, false);
        #pragma unroll
        for (int r = 0; r < 8; r++)
            Sl[r + 8*khalf][kk0 + rloc] = c[r] * scale;
    }
    __syncthreads();
    if (lane < 16){   // row-wise softmax of the 16 x Lk strip
        float mx = -1e30f;
        for (int ci = 0; ci < Lk; ci++) mx = fmaxf(mx, Sl[lane][ci]);
        float sm = 0.0f;
        for (int ci = 0; ci < Lk; ci++){ float ev = __expf(Sl[lane][ci]-mx); Sl[lane][ci] = ev; sm += ev; }
        float inv = 1.0f/sm;
        for (int ci = 0; ci < Lk; ci++) Pl[lane][ci] = (bf16)(Sl[lane][ci]*inv);
        for (int ci = Lk; ci < LKPAD; ci++) Pl[lane][ci] = (bf16)0.0f;
    }
    __syncthreads();
    v8f o0, o1;
    #pragma unroll
    for (int r = 0; r < 8; r++){ o0[r] = 0.0f; o1[r] = 0.0f; }
    int KC = (Lk + 31) >> 5;
    for (int kc = 0; kc < KC; kc++){
        int kk0 = kc*32;
        v16bf ap;
        {
            bf8 lo = *(const bf8*)&Pl[rloc][kk0 + 8*khalf];
            bf8 hi = *(const bf8*)&Pl[rloc][kk0 + 16 + 8*khalf];
            #pragma unroll
            for (int t = 0; t < 8; t++){ ap[t] = lo[t]; ap[t+8] = hi[t]; }
        }
        // V rows beyond Lk are multiplied by zero P-columns: clamp, load unconditionally
        const bf16* vp = V + (size_t)m*Lk*Dd + h*32 + rloc;
        v16bf bv0, bv1;
        {
            #pragma unroll
            for (int t = 0; t < 8; t++){
                size_t k1 = (size_t)min(kk0 + 8*khalf + t, Lk-1)*Dd;
                size_t k2 = (size_t)min(kk0 + 16 + 8*khalf + t, Lk-1)*Dd;
                bv0[t] = vp[k1]; bv0[t+8] = vp[k2];
                bv1[t] = vp[k1 + 16]; bv1[t+8] = vp[k2 + 16];
            }
        }
        o0 = __builtin_amdgcn_wmma_f32_16x16x32_bf16(false, ap, false, bv0, (short)0, o0, false, false);
        o1 = __builtin_amdgcn_wmma_f32_16x16x32_bf16(false, ap, false, bv1, (short)0, o1, false, false);
    }
    #pragma unroll
    for (int r = 0; r < 8; r++){
        int row = q0 + r + 8*khalf;
        if (row < Lq){
            O[((size_t)m*Lq + row)*Dd + h*32 + rloc]      = o0[r];
            O[((size_t)m*Lq + row)*Dd + h*32 + 16 + rloc] = o1[r];
        }
    }
}

// ---------------- LayerNorm over D=128, up to 3 summed inputs; 1 wave/row ----------------
__global__ __launch_bounds__(256)
void ln_kernel(float* __restrict__ out, const float* __restrict__ a,
               const float* __restrict__ b, const float* __restrict__ c, int rows){
    int gw = blockIdx.x*8 + (threadIdx.x >> 5);
    int lane = threadIdx.x & 31;
    if (gw >= rows) return;
    float x[4];
    #pragma unroll
    for (int t = 0; t < 4; t++){
        int d = lane*4 + t;
        float v = a[(size_t)gw*Dd + d];
        if (b) v += b[(size_t)gw*Dd + d];
        if (c) v += c[(size_t)gw*Dd + d];
        x[t] = v;
    }
    float s = x[0]+x[1]+x[2]+x[3];
    #pragma unroll
    for (int off = 16; off > 0; off >>= 1) s += __shfl_xor(s, off, 32);
    float mean = s * (1.0f/128.0f);
    float vs = 0.0f;
    #pragma unroll
    for (int t = 0; t < 4; t++){ float dd = x[t]-mean; vs += dd*dd; }
    #pragma unroll
    for (int off = 16; off > 0; off >>= 1) vs += __shfl_xor(vs, off, 32);
    float rstd = rsqrtf(vs*(1.0f/128.0f) + 1e-5f);
    #pragma unroll
    for (int t = 0; t < 4; t++)
        out[(size_t)gw*Dd + lane*4 + t] = (x[t]-mean)*rstd;
}

// ---------------- naive DFT rfft (192-pt) / irfft+residual ----------------
__global__ __launch_bounds__(128)
void rfft_kernel(float* __restrict__ xr, const float* __restrict__ x){
    __shared__ float cs[Ls], sn[Ls];
    int blk = blockIdx.x; int f = blk % LFn; int bn = blk / LFn;
    int d = threadIdx.x;
    for (int l = d; l < Ls; l += 128){
        float th = 2.0f*PIF*(float)f*(float)l/(float)Ls;
        sincosf(th, &sn[l], &cs[l]);
    }
    __syncthreads();
    float re = 0.0f, im = 0.0f;
    const float* xp = x + (size_t)bn*Ls*Dd + d;
    for (int l = 0; l < Ls; l++){
        float v = xp[(size_t)l*Dd];
        re += v*cs[l]; im -= v*sn[l];
    }
    xr[((size_t)bn*LXn + f)*Dd + d]       = re;
    xr[((size_t)bn*LXn + LFn + f)*Dd + d] = im;
}

__global__ __launch_bounds__(128)
void irfft_add_kernel(float* __restrict__ sf, const float* __restrict__ o, const float* __restrict__ xf){
    __shared__ float cs[LFn], sn[LFn];
    int blk = blockIdx.x; int l = blk % Ls; int bn = blk / Ls;
    int d = threadIdx.x;
    for (int f = d; f < LFn; f += 128){
        float th = 2.0f*PIF*(float)f*(float)l/(float)Ls;
        sincosf(th, &sn[f], &cs[f]);
    }
    __syncthreads();
    const float* ob = o + (size_t)bn*LXn*Dd + d;
    float acc = ob[0];                                        // f = 0 (real)
    acc += ((l & 1) ? -1.0f : 1.0f) * ob[(size_t)96*Dd];      // f = n/2 (real)
    for (int f = 1; f < 96; f++)
        acc += 2.0f*( ob[(size_t)f*Dd]*cs[f] - ob[(size_t)(LFn+f)*Dd]*sn[f] );
    size_t idx = ((size_t)bn*Ls + l)*Dd + d;
    sf[idx] = xf[idx] + acc*(1.0f/(float)Ls);
}

// ---------------- causal-conv input builder: cat[m,l, tap*128+d] = tin[m, l-2+tap, d] ----------------
__global__ void shift3_kernel(float* __restrict__ cat, const float* __restrict__ tin){
    long i = (long)blockIdx.x*256 + threadIdx.x;
    long total = (long)BNn*Ls*3*Dd;
    if (i >= total) return;
    int d = (int)(i % Dd); long r = i / Dd;
    int tap = (int)(r % 3); long r2 = r / 3;
    int l = (int)(r2 % Ls); int bn = (int)(r2 / Ls);
    int ls = l - 2 + tap;
    cat[i] = (ls >= 0) ? tin[((size_t)bn*Ls + ls)*Dd + d] : 0.0f;
}

// ---------------- moving average split ----------------
__global__ __launch_bounds__(128)
void movavg_kernel(float* __restrict__ xfreq, float* __restrict__ xtrend, const float* __restrict__ yy){
    int bl = blockIdx.x; int l = bl % Ls; int bn = bl / Ls;
    int d = threadIdx.x;
    float s = 0.0f;
    #pragma unroll
    for (int i = -12; i <= 12; i++){
        int li = min(max(l+i, 0), Ls-1);
        s += yy[((size_t)bn*Ls + li)*Dd + d];
    }
    float ma = s*(1.0f/25.0f);
    size_t idx = ((size_t)bn*Ls + l)*Dd + d;
    xtrend[idx] = ma;
    xfreq[idx]  = yy[idx] - ma;
}

__global__ void add2_kernel(float* __restrict__ z, const float* __restrict__ a, const float* __restrict__ b){
    long i = (long)blockIdx.x*256 + threadIdx.x;
    if (i >= TOKN) return;
    z[i] = a[i] + b[i];
}

// ---------------- final projection to 1 channel, take last PRED steps ----------------
__global__ __launch_bounds__(256)
void proj_kernel(float* __restrict__ out, const float* __restrict__ z,
                 const float* __restrict__ wp, const float* __restrict__ bp){
    int gw = blockIdx.x*8 + (threadIdx.x >> 5);
    int lane = threadIdx.x & 31;
    if (gw >= BNn*PREDn) return;
    int bn = gw / PREDn, p = gw % PREDn;
    const float* zp = z + ((size_t)bn*Ls + (SEQn + p))*Dd;
    float s = 0.0f;
    #pragma unroll
    for (int t = 0; t < 4; t++){ int d = lane*4+t; s += zp[d]*wp[d]; }
    #pragma unroll
    for (int off = 16; off > 0; off >>= 1) s += __shfl_xor(s, off, 32);
    if (lane == 0) out[gw] = s + bp[0];
}

// ---------------- host orchestration ----------------
extern "C" void kernel_launch(void* const* d_in, const int* in_sizes, int n_in,
                              void* d_out, int out_size, void* d_ws, size_t ws_size,
                              hipStream_t stream){
    (void)in_sizes; (void)n_in; (void)out_size; (void)ws_size;
    const float* nodes  = (const float*)d_in[0];
    const float* xmark  = (const float*)d_in[1];
    const float* edges  = (const float*)d_in[2];
    const float* wcf    = (const float*)d_in[3];
    const float* wct    = (const float*)d_in[4];
    const float* wtemp  = (const float*)d_in[5];
    const float* wedge  = (const float*)d_in[6];
    const float* bedge  = (const float*)d_in[7];
    const float* wp     = (const float*)d_in[26];
    const float* bp     = (const float*)d_in[27];
    const int*   snd    = (const int*)d_in[28];
    const int*   rcv    = (const int*)d_in[29];
    float* out = (float*)d_out;

    char* wsp = (char*)d_ws;
    size_t off = 0;
    auto allocF = [&](long n)->float*{ float* p = (float*)(wsp + off); off += (size_t)n*4; off = (off + 255) & ~(size_t)255; return p; };
    auto allocB = [&](long n)->bf16*{ bf16* p = (bf16*)(wsp + off); off += (size_t)n*2; off = (off + 255) & ~(size_t)255; return p; };

    // bf16 weight arena
    const int DD = Dd*Dd, DF = Dd*DFFn, LQ3 = 3*Dd*Dd;
    bf16 *eqb = allocB(DD), *ekb = allocB(DD), *evb = allocB(DD), *eob = allocB(DD);
    bf16 *eff1b = allocB(DF), *eff2b = allocB(DF);
    bf16 *fqb = allocB(DD), *fkb = allocB(DD), *fvb = allocB(DD), *fob = allocB(DD);
    bf16 *lqb = allocB(LQ3), *lkb = allocB(LQ3), *lvb = allocB(DD), *lob = allocB(DD);
    bf16 *nff1b = allocB(DF), *nff2b = allocB(DF), *m1b = allocB(DF), *m2b = allocB(DF);

    // activation arena
    float* freqd  = allocF((long)BNn*Ls*Jj*Cc);
    float* trendd = allocF((long)BNn*Ls*Cc);
    float* xfreq  = allocF(TOKN);
    float* xtrend = allocF(TOKN);
    float* eb0    = allocF(TOKE);
    float* eb1    = allocF(TOKE);
    float* cnt    = allocF(BNn);
    float* ein    = allocF(TOKE);
    bf16*  qbh    = allocB(TOKE);   // attention Q/K/V live in bf16
    bf16*  kbh    = allocB(TOKE);
    bf16*  vbh    = allocB(TOKE);
    float* ao     = allocF(TOKE);
    float* yb     = allocF(TOKE);
    float* tmp1   = allocF(TOKE);
    float* mid    = allocF(FFNE);   // also reused for shift3 concat + node FFN mid
    float* tin    = allocF(TOKN);
    float* xr     = allocF(TOKF);
    float* sf     = allocF(TOKN);
    float* yy     = allocF(TOKN);
    float* yy2    = allocF(TOKN);

    auto cvt = [&](bf16* dst, int idx, int n){
        cvt_bf16_kernel<<<(n+255)/256, 256, 0, stream>>>(dst, (const float*)d_in[idx], n);
    };
    cvt(eqb,8,DD);  cvt(ekb,9,DD);  cvt(evb,10,DD); cvt(eob,11,DD);
    cvt(eff1b,12,DF); cvt(eff2b,13,DF);
    cvt(fqb,14,DD); cvt(fkb,15,DD); cvt(fvb,16,DD); cvt(fob,17,DD);
    cvt(lqb,18,LQ3); cvt(lkb,19,LQ3); cvt(lvb,20,DD); cvt(lob,21,DD);
    cvt(nff1b,22,DF); cvt(nff2b,23,DF); cvt(m1b,24,DF); cvt(m2b,25,DF);

    auto gemm = [&](float* Y, const float* A, const bf16* W, const float* R,
                    int M, int K, int N, int act){
        dim3 g((M+63)/64, N/128);
        gemm_bf16_kernel<<<g, 256, 0, stream>>>(Y, (bf16*)nullptr, A, W, R, M, K, N, act);
    };
    auto gemmb = [&](bf16* Ybf, const float* A, const bf16* W, int M, int K, int N){
        dim3 g((M+63)/64, N/128);
        gemm_bf16_kernel<<<g, 256, 0, stream>>>((float*)nullptr, Ybf, A, W, (const float*)nullptr, M, K, N, 0);
    };
    auto attn = [&](float* O, const bf16* Qp, const bf16* Kp, const bf16* Vp,
                    int M, int Lq, int Lk){
        int QT = (Lq+15)/16;
        attn_kernel<<<M*Hh*QT, 32, 0, stream>>>(O, Qp, Kp, Vp, M, Lq, Lk);
    };
    auto ln = [&](float* o, const float* a, const float* b, const float* c, int rows){
        ln_kernel<<<(rows+7)/8, 256, 0, stream>>>(o, a, b, c, rows);
    };
    auto zero = [&](float* p, long n){
        zero_kernel<<<(int)((n+255)/256), 256, 0, stream>>>(p, n);
    };

    // ---- embeddings ----
    wavelet_kernel<<<BNn, 256, 0, stream>>>(nodes, freqd, trendd);
    embed_freq_kernel<<<BNn*Ls, 128, 0, stream>>>(xfreq, freqd, wcf);
    embed_trend_kernel<<<BNn*Ls, 128, 0, stream>>>(xtrend, trendd, wct, xmark, wtemp);
    edge_init_kernel<<<(int)((TOKE+255)/256), 256, 0, stream>>>(eb0, edges, wedge, bedge);
    zero(cnt, BNn);
    count_kernel<<<(Ee+255)/256, 256, 0, stream>>>(cnt, rcv);

    const int ME = Ee*Ls;          // 153600 edge tokens
    const int MN = BNn*Ls;         // 38400  node tokens
    const int MF = BNn*LXn;        // 38800  spectral tokens

    float* ecur = eb0;
    float* enext = eb1;
    for (int layer = 0; layer < 2; layer++){
        // -------- edge transformer --------
        gather_ein_kernel<<<(int)((TOKE+255)/256), 256, 0, stream>>>(ein, ecur, xtrend, snd, rcv);
        gemmb(qbh, ein, eqb, ME, Dd, Dd);
        gemmb(kbh, ein, ekb, ME, Dd, Dd);
        gemmb(vbh, ein, evb, ME, Dd, Dd);
        attn(ao, qbh, kbh, vbh, Ee, Ls, Ls);
        gemm(tmp1, ao, eob, nullptr, ME, Dd, Dd, 0);
        ln(yb, ein, tmp1, nullptr, ME);
        gemm(mid, yb, eff1b, nullptr, ME, Dd, DFFn, 1);          // gelu
        gemm(tmp1, mid, eff2b, nullptr, ME, DFFn, Dd, 0);
        ln(enext, yb, tmp1, nullptr, ME);                        // e_out

        // -------- node aggregation --------
        zero(tin, TOKN);
        agg_kernel<<<(int)((TOKE+255)/256), 256, 0, stream>>>(tin, enext, rcv);
        tin_finish_kernel<<<(int)((TOKN+255)/256), 256, 0, stream>>>(tin, xtrend, cnt);

        // -------- FFT attention branch --------
        rfft_kernel<<<BNn*LFn, 128, 0, stream>>>(xr, xfreq);
        gemmb(qbh, xr, fqb, MF, Dd, Dd);
        gemmb(kbh, xr, fkb, MF, Dd, Dd);
        gemmb(vbh, xr, fvb, MF, Dd, Dd);
        attn(ao, qbh, kbh, vbh, BNn, LXn, LXn);
        gemm(tmp1, ao, fob, nullptr, MF, Dd, Dd, 0);
        irfft_add_kernel<<<BNn*Ls, 128, 0, stream>>>(sf, tmp1, xfreq);

        // -------- log-sparse branch (causal conv q/k as K=384 GEMM) --------
        shift3_kernel<<<(int)(((long)MN*3*Dd+255)/256), 256, 0, stream>>>(mid, tin);
        gemmb(qbh, mid, lqb, MN, 3*Dd, Dd);
        gemmb(kbh, mid, lkb, MN, 3*Dd, Dd);
        gemmb(vbh, tin, lvb, MN, Dd, Dd);
        attn(ao, qbh, kbh, vbh, BNn, Ls, Ls);
        gemm(tmp1, ao, lob, nullptr, MN, Dd, Dd, 0);
        ln(yy, sf, tin, tmp1, MN);                               // LN(s_f + t_in + attn)
        gemm(mid, yy, nff1b, nullptr, MN, Dd, DFFn, 1);          // gelu
        gemm(tmp1, mid, nff2b, nullptr, MN, DFFn, Dd, 0);
        ln(yy2, yy, tmp1, nullptr, MN);
        movavg_kernel<<<BNn*Ls, 128, 0, stream>>>(xfreq, xtrend, yy2);

        float* t = ecur; ecur = enext; enext = t;                // e <- e_out
    }

    // ---- head ----
    add2_kernel<<<(int)((TOKN+255)/256), 256, 0, stream>>>(sf, xfreq, xtrend);  // z
    gemm(mid, sf, m1b, nullptr, MN, Dd, DFFn, 1);                // gelu(z @ m1)
    gemm(yy, mid, m2b, sf, MN, DFFn, Dd, 0);                     // z + ... @ m2
    proj_kernel<<<(BNn*PREDn+7)/8, 256, 0, stream>>>(out, yy, wp, bp);
}
